// HeteroEdgeAwareTransformer_41180146434512
// MI455X (gfx1250) — matrix operand order repaired and checked by reading, not compiled, not run on hardware
//
#include <hip/hip_runtime.h>
#include <hip/hip_bf16.h>
#include <math.h>

// ---------------------------------------------------------------------------
// HeteroEdgeAwareTransformer forward for MI455X (gfx1250, wave32, WMMA).
// Dense projections -> v_wmma_f32_16x16x32_f16 (f16 in, f32 accum).
// Edge phase -> on-the-fly edge projection from LDS-resident `we`,
// segment-max via encoded-uint atomicMax, scatter via global f32 atomics.
// ---------------------------------------------------------------------------

#define NN   50000
#define EE   400000
#define HID  128
#define HH   256        // HEADS * HIDDEN
#define INSCALE 0.08838834764831843f   // 1/sqrt(128)

typedef __attribute__((ext_vector_type(16))) _Float16 v16h;
typedef __attribute__((ext_vector_type(8)))  _Float16 v8h;
typedef __attribute__((ext_vector_type(8)))  float    v8f;

static __device__ inline v16h frag16(const _Float16* p0, const _Float16* p1) {
  v8h lo = *(const v8h*)p0;
  v8h hi = *(const v8h*)p1;
  v16h r;
#pragma unroll
  for (int i = 0; i < 8; ++i) { r[i] = lo[i]; r[i + 8] = hi[i]; }
  return r;
}

// order-preserving float<->uint encoding for atomicMax-based segment max
static __device__ inline unsigned fenc(float x) {
  int i = __float_as_int(x);
  return (i >= 0) ? ((unsigned)i ^ 0x80000000u) : ~(unsigned)i;
}
static __device__ inline float fdec(unsigned u) {
  unsigned i = (u & 0x80000000u) ? (u ^ 0x80000000u) : ~u;
  return __int_as_float((int)i);
}

// ---- weight prep: W[K,Nc] f32 -> Wt[Nc,Kp] f16 (transposed, zero-padded K) --
__global__ void wt_cvt(const float* __restrict__ W, _Float16* __restrict__ Wt,
                       int K, int Nc, int Kp) {
  int idx = blockIdx.x * blockDim.x + threadIdx.x;
  if (idx >= Nc * Kp) return;
  int n = idx / Kp, k = idx % Kp;
  Wt[idx] = (k < K) ? (_Float16)W[(size_t)k * Nc + n] : (_Float16)0.0f;
}

// ---- activation prep: A[M,K] f32 -> Ah[M,Kp] f16 (zero-padded K) -----------
__global__ void cvt_pad(const float* __restrict__ A, _Float16* __restrict__ Ah,
                        int M, int K, int Kp) {
  int idx = blockIdx.x * blockDim.x + threadIdx.x;
  if (idx >= M * Kp) return;
  int r = idx / Kp, k = idx % Kp;
  Ah[idx] = (k < K) ? (_Float16)A[(size_t)r * K + k] : (_Float16)0.0f;
}

// ---- tiled WMMA GEMM: C[M,Nc] = act(Ah[M,Kp] x Bt[Nc,Kp]^T + bias) ---------
// one wave -> one 16x16 output tile; K-loop in steps of 32
__global__ void gemm_f16_wmma(const _Float16* __restrict__ Ah,
                              const _Float16* __restrict__ Bt,
                              const float* __restrict__ bias,
                              float* __restrict__ C,
                              int M, int Nc, int Kp, int act) {
  int wid  = (blockIdx.x * blockDim.x + threadIdx.x) >> 5;
  int lane = threadIdx.x & 31;
  int ntiles = Nc >> 4;
  int total  = (M >> 4) * ntiles;
  if (wid >= total) return;
  int nt = wid % ntiles, mt = wid / ntiles;
  int m0 = mt << 4, n0 = nt << 4;
  int r = lane & 15, hi = lane >> 4;

  const _Float16* arow = Ah + (size_t)(m0 + r) * Kp;
  const _Float16* brow = Bt + (size_t)(n0 + r) * Kp;

  v8f acc = {};
  for (int k0 = 0; k0 < Kp; k0 += 32) {
    v16h a = frag16(arow + k0 + hi * 8,  arow + k0 + 16 + hi * 8);
    v16h b = frag16(brow + k0 + hi * 8,  brow + k0 + 16 + hi * 8);
    acc = __builtin_amdgcn_wmma_f32_16x16x32_f16(
        false, a, false, b, (short)0, acc, false, false);
  }
  float bn = bias[n0 + r];
#pragma unroll
  for (int i = 0; i < 8; ++i) {
    float v = acc[i] + bn;
    if (act == 1) v = v > 0.f ? v : 0.f;
    C[(size_t)(m0 + hi * 8 + i) * Nc + n0 + r] = v;
  }
}

// ---- edge feature projection: EA[E,4] = attr[E,4] @ ep_w + ep_b ------------
__global__ void edge_proj(const float* __restrict__ attr,
                          const float* __restrict__ epw,
                          const float* __restrict__ epb,
                          float* __restrict__ EA, int E_) {
  int e = blockIdx.x * blockDim.x + threadIdx.x;
  if (e >= E_) return;
  float a0 = attr[e * 4 + 0], a1 = attr[e * 4 + 1];
  float a2 = attr[e * 4 + 2], a3 = attr[e * 4 + 3];
#pragma unroll
  for (int j = 0; j < 4; ++j)
    EA[e * 4 + j] = a0 * epw[j] + a1 * epw[4 + j] + a2 * epw[8 + j]
                  + a3 * epw[12 + j] + epb[j];
}

// ---- in-place LayerNorm over 128 channels, one wave per row ----------------
__global__ void ln_inplace(float* __restrict__ X, const float* __restrict__ g,
                           const float* __restrict__ b, int n) {
  int wid  = (blockIdx.x * blockDim.x + threadIdx.x) >> 5;
  int lane = threadIdx.x & 31;
  if (wid >= n) return;
  float t[4], sum = 0.f, sq = 0.f;
#pragma unroll
  for (int j = 0; j < 4; ++j) {
    int c = lane + 32 * j;
    t[j] = X[(size_t)wid * HID + c];
    sum += t[j]; sq += t[j] * t[j];
  }
#pragma unroll
  for (int m = 16; m >= 1; m >>= 1) { sum += __shfl_xor(sum, m); sq += __shfl_xor(sq, m); }
  float mu = sum * (1.f / HID);
  float var = sq * (1.f / HID) - mu * mu;
  float rs = rsqrtf(var + 1e-5f);
#pragma unroll
  for (int j = 0; j < 4; ++j) {
    int c = lane + 32 * j;
    X[(size_t)wid * HID + c] = (t[j] - mu) * rs * g[c] + b[c];
  }
}

// ---- per-layer segment-state init ------------------------------------------
__global__ void init_seg(unsigned* __restrict__ ME, float* __restrict__ S,
                         float* __restrict__ AG, int n) {
  int idx = blockIdx.x * blockDim.x + threadIdx.x;
  if (idx < n * HH) AG[idx] = 0.f;
  if (idx < n * 2) { S[idx] = 0.f; ME[idx] = 0x007FFFFFu; }  // fenc(-inf)
}

// ---- edge pass 1: alpha + segment max. one wave per edge -------------------
__global__ void edge_alpha(const int* __restrict__ ei,
                           const float* __restrict__ Q,
                           const float* __restrict__ Kk,
                           const float* __restrict__ EA,
                           const float* __restrict__ we,   // [4,256]
                           float* __restrict__ AL,
                           unsigned* __restrict__ ME, int E_) {
  __shared__ float sWe[4 * HH];
  for (int i = threadIdx.x; i < 4 * HH; i += blockDim.x) sWe[i] = we[i];
  __syncthreads();
  int wid  = (blockIdx.x * blockDim.x + threadIdx.x) >> 5;
  int lane = threadIdx.x & 31;
  if (wid >= E_) return;
  int src = ei[wid], dst = ei[E_ + wid];
  int h = lane >> 4, li = lane & 15;
  int cb = h * HID + li * 8;
  float a0 = EA[wid * 4 + 0], a1 = EA[wid * 4 + 1];
  float a2 = EA[wid * 4 + 2], a3 = EA[wid * 4 + 3];
  const float* qp = Q  + (size_t)dst * HH + cb;
  const float* kp = Kk + (size_t)src * HH + cb;
  const float* wp = sWe + cb;
  float p = 0.f;
#pragma unroll
  for (int j = 0; j < 8; ++j) {
    float ev = a0 * wp[j] + a1 * wp[HH + j] + a2 * wp[2 * HH + j] + a3 * wp[3 * HH + j];
    p += qp[j] * (kp[j] + ev);
  }
#pragma unroll
  for (int m = 8; m >= 1; m >>= 1) p += __shfl_xor(p, m);   // within 16-lane half
  float al = p * INSCALE;
  if (li == 0) {
    AL[wid * 2 + h] = al;
    atomicMax(ME + (size_t)dst * 2 + h, fenc(al));
  }
}

// ---- edge pass 2: exp + scatter (unnormalized; divide by S at node) --------
__global__ void edge_scatter(const int* __restrict__ ei,
                             const float* __restrict__ Vv,
                             const float* __restrict__ EA,
                             const float* __restrict__ we,
                             const float* __restrict__ AL,
                             const unsigned* __restrict__ ME,
                             float* __restrict__ S,
                             float* __restrict__ AG, int E_) {
  __shared__ float sWe[4 * HH];
  for (int i = threadIdx.x; i < 4 * HH; i += blockDim.x) sWe[i] = we[i];
  __syncthreads();
  int wid  = (blockIdx.x * blockDim.x + threadIdx.x) >> 5;
  int lane = threadIdx.x & 31;
  if (wid >= E_) return;
  int src = ei[wid], dst = ei[E_ + wid];
  int h = lane >> 4, li = lane & 15;
  int cb = h * HID + li * 8;
  float m = fdec(ME[(size_t)dst * 2 + h]);
  float a = __expf(AL[wid * 2 + h] - m);
  if (li == 0) atomicAdd(S + (size_t)dst * 2 + h, a);
  float a0 = EA[wid * 4 + 0], a1 = EA[wid * 4 + 1];
  float a2 = EA[wid * 4 + 2], a3 = EA[wid * 4 + 3];
  const float* vp = Vv + (size_t)src * HH + cb;
  const float* wp = sWe + cb;
#pragma unroll
  for (int j = 0; j < 8; ++j) {
    float ev = a0 * wp[j] + a1 * wp[HH + j] + a2 * wp[2 * HH + j] + a3 * wp[3 * HH + j];
    atomicAdd(AG + (size_t)dst * HH + cb + j, (vp[j] + ev) * a);
  }
}

// ---- node update: normalize heads, mean, +skip, ELU, residual, LN ----------
__global__ void node_update(float* __restrict__ X,
                            const float* __restrict__ AG,
                            const float* __restrict__ S,
                            const float* __restrict__ SK,
                            const float* __restrict__ lng,
                            const float* __restrict__ lnb, int n) {
  int wid  = (blockIdx.x * blockDim.x + threadIdx.x) >> 5;
  int lane = threadIdx.x & 31;
  if (wid >= n) return;
  float inv0 = 1.f / (S[(size_t)wid * 2 + 0] + 1e-16f);
  float inv1 = 1.f / (S[(size_t)wid * 2 + 1] + 1e-16f);
  float t[4], sum = 0.f, sq = 0.f;
#pragma unroll
  for (int j = 0; j < 4; ++j) {
    int c = lane + 32 * j;
    float g = 0.5f * (AG[(size_t)wid * HH + c] * inv0 +
                      AG[(size_t)wid * HH + HID + c] * inv1);
    float h = g + SK[(size_t)wid * HID + c];
    h = h > 0.f ? h : (__expf(h) - 1.f);                 // ELU
    float tv = X[(size_t)wid * HID + c] + h;             // residual
    t[j] = tv; sum += tv; sq += tv * tv;
  }
#pragma unroll
  for (int m = 16; m >= 1; m >>= 1) { sum += __shfl_xor(sum, m); sq += __shfl_xor(sq, m); }
  float mu = sum * (1.f / HID);
  float var = sq * (1.f / HID) - mu * mu;
  float rs = rsqrtf(var + 1e-5f);
#pragma unroll
  for (int j = 0; j < 4; ++j) {
    int c = lane + 32 * j;
    X[(size_t)wid * HID + c] = (t[j] - mu) * rs * lng[c] + lnb[c];
  }
}

// ---- output head: out[n] = H1[n,:] . h2_w + h2_b ---------------------------
__global__ void head_out(const float* __restrict__ H1b,
                         const float* __restrict__ w2,
                         const float* __restrict__ b2,
                         float* __restrict__ out, int n) {
  int wid  = (blockIdx.x * blockDim.x + threadIdx.x) >> 5;
  int lane = threadIdx.x & 31;
  if (wid >= n) return;
  float p = 0.f;
#pragma unroll
  for (int j = 0; j < 4; ++j) {
    int c = lane + 32 * j;
    p += H1b[(size_t)wid * HID + c] * w2[c];
  }
#pragma unroll
  for (int m = 16; m >= 1; m >>= 1) p += __shfl_xor(p, m);
  if (lane == 0) out[wid] = p + b2[0];
}

// ---------------------------------------------------------------------------
extern "C" void kernel_launch(void* const* d_in, const int* in_sizes, int n_in,
                              void* d_out, int out_size, void* d_ws, size_t ws_size,
                              hipStream_t stream) {
  // input index map (setup_inputs() dict/list insertion order):
  // 0:x 1:edge_index 2:edge_attr 3:in_w 4:in_b 5:ep_w 6:ep_b 7:ln0_g 8:ln0_b
  // 9:h1_w 10:h1_b 11:h2_w 12:h2_b  then per layer l (base=13+11*l):
  // +0:wq +1:bq +2:wk +3:bk +4:wv +5:bv +6:we +7:wskip +8:bskip +9:ln_g +10:ln_b
  const float* x_in   = (const float*)d_in[0];
  const int*   eidx   = (const int*)d_in[1];
  const float* eattr  = (const float*)d_in[2];
  const float* in_w   = (const float*)d_in[3];
  const float* in_b   = (const float*)d_in[4];
  const float* ep_w   = (const float*)d_in[5];
  const float* ep_b   = (const float*)d_in[6];
  const float* ln0_g  = (const float*)d_in[7];
  const float* ln0_b  = (const float*)d_in[8];
  const float* h1_w   = (const float*)d_in[9];
  const float* h1_b   = (const float*)d_in[10];
  const float* h2_w   = (const float*)d_in[11];
  const float* h2_b   = (const float*)d_in[12];

  // ---- workspace carving ----
  char* base = (char*)d_ws;
  size_t off = 0;
  auto carve = [&](size_t bytes) -> char* {
    char* p = base + off;
    off += (bytes + 255) & ~(size_t)255;
    return p;
  };
  float*    X    = (float*)carve((size_t)NN * HID * 4);
  _Float16* Xh   = (_Float16*)carve((size_t)NN * HID * 2);
  float*    Qb   = (float*)carve((size_t)NN * HH * 4);
  float*    Kb   = (float*)carve((size_t)NN * HH * 4);
  float*    Vb   = (float*)carve((size_t)NN * HH * 4);
  float*    SK   = (float*)carve((size_t)NN * HID * 4);
  float*    EA   = (float*)carve((size_t)EE * 4 * 4);
  float*    AL   = (float*)carve((size_t)EE * 2 * 4);
  unsigned* ME   = (unsigned*)carve((size_t)NN * 2 * 4);
  float*    S    = (float*)carve((size_t)NN * 2 * 4);
  float*    AG   = (float*)carve((size_t)NN * HH * 4);
  float*    H1b  = (float*)carve((size_t)NN * HID * 4);
  _Float16* WT_IN = (_Float16*)carve((size_t)128 * 32 * 2);
  _Float16* WT_H1 = (_Float16*)carve((size_t)128 * 128 * 2);
  _Float16* WT_Q[3]; _Float16* WT_K[3]; _Float16* WT_V[3]; _Float16* WT_S[3];
  for (int l = 0; l < 3; ++l) {
    WT_Q[l] = (_Float16*)carve((size_t)256 * 128 * 2);
    WT_K[l] = (_Float16*)carve((size_t)256 * 128 * 2);
    WT_V[l] = (_Float16*)carve((size_t)256 * 128 * 2);
    WT_S[l] = (_Float16*)carve((size_t)128 * 128 * 2);
  }
  (void)ws_size; (void)n_in; (void)in_sizes; (void)out_size;

  const int TB = 256;
  auto blk = [](long threads) { return (int)((threads + 255) / 256); };

  // ---- weight conversion (transpose + f16 + K-pad) ----
  wt_cvt<<<blk(128 * 32), TB, 0, stream>>>(in_w, WT_IN, 16, 128, 32);
  wt_cvt<<<blk(128 * 128), TB, 0, stream>>>(h1_w, WT_H1, 128, 128, 128);
  for (int l = 0; l < 3; ++l) {
    int b0 = 13 + 11 * l;
    wt_cvt<<<blk(256 * 128), TB, 0, stream>>>((const float*)d_in[b0 + 0], WT_Q[l], 128, 256, 128);
    wt_cvt<<<blk(256 * 128), TB, 0, stream>>>((const float*)d_in[b0 + 2], WT_K[l], 128, 256, 128);
    wt_cvt<<<blk(256 * 128), TB, 0, stream>>>((const float*)d_in[b0 + 4], WT_V[l], 128, 256, 128);
    wt_cvt<<<blk(128 * 128), TB, 0, stream>>>((const float*)d_in[b0 + 7], WT_S[l], 128, 128, 128);
  }

  // ---- edge feature projection ----
  edge_proj<<<blk(EE), TB, 0, stream>>>(eattr, ep_w, ep_b, EA, EE);

  // ---- input projection + relu, then LN0 ----
  cvt_pad<<<blk((long)NN * 32), TB, 0, stream>>>(x_in, Xh, NN, 16, 32);
  gemm_f16_wmma<<<blk((long)(NN / 16) * (128 / 16) * 32), TB, 0, stream>>>(
      Xh, WT_IN, in_b, X, NN, 128, 32, 1);
  ln_inplace<<<blk((long)NN * 32), TB, 0, stream>>>(X, ln0_g, ln0_b, NN);

  // ---- transformer layers ----
  for (int l = 0; l < 3; ++l) {
    int b0 = 13 + 11 * l;
    const float* bq    = (const float*)d_in[b0 + 1];
    const float* bk    = (const float*)d_in[b0 + 3];
    const float* bv    = (const float*)d_in[b0 + 5];
    const float* we    = (const float*)d_in[b0 + 6];
    const float* bskip = (const float*)d_in[b0 + 8];
    const float* lng   = (const float*)d_in[b0 + 9];
    const float* lnb   = (const float*)d_in[b0 + 10];

    cvt_pad<<<blk((long)NN * HID), TB, 0, stream>>>(X, Xh, NN, HID, HID);

    long gw256 = (long)(NN / 16) * (HH / 16) * 32;
    long gw128 = (long)(NN / 16) * (HID / 16) * 32;
    gemm_f16_wmma<<<blk(gw256), TB, 0, stream>>>(Xh, WT_Q[l], bq, Qb, NN, HH, HID, 0);
    gemm_f16_wmma<<<blk(gw256), TB, 0, stream>>>(Xh, WT_K[l], bk, Kb, NN, HH, HID, 0);
    gemm_f16_wmma<<<blk(gw256), TB, 0, stream>>>(Xh, WT_V[l], bv, Vb, NN, HH, HID, 0);
    gemm_f16_wmma<<<blk(gw128), TB, 0, stream>>>(Xh, WT_S[l], bskip, SK, NN, HID, HID, 0);

    init_seg<<<blk((long)NN * HH), TB, 0, stream>>>(ME, S, AG, NN);
    edge_alpha<<<blk((long)EE * 32), TB, 0, stream>>>(eidx, Qb, Kb, EA, we, AL, ME, EE);
    edge_scatter<<<blk((long)EE * 32), TB, 0, stream>>>(eidx, Vb, EA, we, AL, ME, S, AG, EE);
    node_update<<<blk((long)NN * 32), TB, 0, stream>>>(X, AG, S, SK, lng, lnb, NN);
  }

  // ---- head MLP ----
  cvt_pad<<<blk((long)NN * HID), TB, 0, stream>>>(X, Xh, NN, HID, HID);
  gemm_f16_wmma<<<blk((long)(NN / 16) * (HID / 16) * 32), TB, 0, stream>>>(
      Xh, WT_H1, h1_b, H1b, NN, HID, HID, 1);
  head_out<<<blk((long)NN * 32), TB, 0, stream>>>(H1b, h2_w, h2_b, (float*)d_out, NN);
}
